// TransformerBlock_1486058684452
// MI455X (gfx1250) — compile-verified, matrix operand order checked by reading, not run on hardware
//
#include <hip/hip_runtime.h>

// ---------------------------------------------------------------------------
// MI455X (gfx1250, wave32) transformer block.
// Compute-bound on bf16 WMMA (~1.03 TFLOP vs ~50us of HBM traffic at 23.3TB/s
// once weights are bf16; each bf16 weight matrix fits in the 192MB L2).
// All matrix math uses v_wmma_f32_16x16x32_bf16 with f32 accumulation.
// GEMMs stage A/B tiles through LDS with GLOBAL_LOAD_ASYNC_TO_LDS_B128
// (ASYNCcnt, double buffered) so the WGP$ only sees unique tile data.
// Workspace requirement: ~620 MB (bf16 weights + activations).
// ---------------------------------------------------------------------------

typedef __attribute__((ext_vector_type(8)))  __bf16 v8bf;
typedef __attribute__((ext_vector_type(16))) __bf16 v16bf;
typedef __attribute__((ext_vector_type(8)))  float  v8f;

constexpr int DIM    = 4096;
constexpr int SEQ    = 2048;
constexpr int CACHE  = 4096;
constexpr int NH     = 32;
constexpr int NKV    = 8;
constexpr int HD     = 128;
constexpr int KVD    = NKV * HD;     // 1024
constexpr int HIDDEN = 14336;

#define DEVFN static __device__ __forceinline__

DEVFN unsigned short f2bf(float f) {
    unsigned int u = __float_as_uint(f);
    u += 0x7fffu + ((u >> 16) & 1u);       // round-to-nearest-even
    return (unsigned short)(u >> 16);
}
DEVFN float bf2f(unsigned short h) {
    return __uint_as_float(((unsigned int)h) << 16);
}

// Load one 16-bf16 WMMA fragment for this lane: K-chunk [0,8) at p and
// K-chunk [16,24) at p+16 (lane>>4 already folded into p by the caller).
DEVFN v16bf load_frag16(const unsigned short* p) {
    v8bf lo = *reinterpret_cast<const v8bf*>(p);
    v8bf hi = *reinterpret_cast<const v8bf*>(p + 16);
    return __builtin_shufflevector(lo, hi, 0, 1, 2, 3, 4, 5, 6, 7,
                                   8, 9, 10, 11, 12, 13, 14, 15);
}

DEVFN v8f wmma_bf16(v16bf a, v16bf b, v8f c) {
    return __builtin_amdgcn_wmma_f32_16x16x32_bf16(
        /*neg_a=*/false, a, /*neg_b=*/false, b,
        /*c_mod=*/(short)0, c, /*reuse_a=*/false, /*reuse_b=*/false);
}

// Async copy of one 128x32 bf16 tile (rows from gbase, k-cols [k0,k0+32))
// into LDS (row stride RS elems).  256 threads -> each issues two
// GLOBAL_LOAD_ASYNC_TO_LDS_B128 ops; tracked on ASYNCcnt.
constexpr int RS = 40;                    // 32 + 8 pad elems (80B row stride)

DEVFN void async_copy_tile(const unsigned short* gbase, unsigned short* lds,
                           int K, int k0, int tid) {
    int row = tid >> 1;
    int ch  = (tid & 1) * 16;             // element column 0 or 16
    const unsigned short* g = gbase + (size_t)row * K + k0 + ch;
    unsigned int l =
        (unsigned int)(unsigned long long)(lds + row * RS + ch);
    unsigned long long ga = (unsigned long long)g;
    asm volatile("global_load_async_to_lds_b128 %0, %1, off"
                 :: "v"(l), "v"(ga) : "memory");
    asm volatile("global_load_async_to_lds_b128 %0, %1, off"
                 :: "v"(l + 16u), "v"(ga + 16ull) : "memory");
}

// ---------------------------------------------------------------------------
// Weight convert: fp32 (K rows x N cols) -> bf16 transposed (N rows x K cols).
// B fragments then become two contiguous 16B loads per lane.
// ---------------------------------------------------------------------------
__global__ void transpose_convert_kernel(const float* __restrict__ w,
                                         unsigned short* __restrict__ wT,
                                         int K, int N) {
    __shared__ float tile[32][33];
    int n0 = blockIdx.x * 32, k0 = blockIdx.y * 32;
    int tx = threadIdx.x, ty = threadIdx.y;   // 32 x 8
#pragma unroll
    for (int i = 0; i < 4; ++i) {
        int k = ty + i * 8;
        tile[k][tx] = w[(size_t)(k0 + k) * N + (n0 + tx)];
    }
    __syncthreads();
#pragma unroll
    for (int i = 0; i < 4; ++i) {
        int n = ty + i * 8;
        wT[(size_t)(n0 + n) * K + (k0 + tx)] = f2bf(tile[tx][n]);
    }
}

// ---------------------------------------------------------------------------
// RMSNorm: fp32 row -> bf16 row.  One 256-thread block per row (DIM=4096).
// ---------------------------------------------------------------------------
__global__ __launch_bounds__(256) void rmsnorm_bf16_kernel(
    const float* __restrict__ x, const float* __restrict__ w,
    unsigned short* __restrict__ out) {
    int row = blockIdx.x, t = threadIdx.x;
    const float* xr = x + (size_t)row * DIM;
    float v[16];
    float ss = 0.f;
#pragma unroll
    for (int i = 0; i < 16; ++i) {
        v[i] = xr[t + i * 256];
        ss += v[i] * v[i];
    }
#pragma unroll
    for (int off = 16; off > 0; off >>= 1) ss += __shfl_xor(ss, off, 32);
    __shared__ float sred[8];
    if ((t & 31) == 0) sred[t >> 5] = ss;
    __syncthreads();
    float tot = 0.f;
#pragma unroll
    for (int i = 0; i < 8; ++i) tot += sred[i];
    float scale = rsqrtf(tot / (float)DIM + 1e-6f);
#pragma unroll
    for (int i = 0; i < 16; ++i)
        out[(size_t)row * DIM + t + i * 256] = f2bf(v[i] * scale * w[t + i * 256]);
}

// ---------------------------------------------------------------------------
// WMMA bf16 GEMM: C[M,N] = A[M,K] * B^T[N,K]   (B stored N-major, bf16).
// Block = 256 threads (8 waves), BM=BN=128; wave tile 64x32 = 8 accum tiles.
// A/B 128x32 tiles double-buffered in LDS via async-to-LDS copies.
// Epilogues: bf16 store / silu(acc0)*acc1 bf16 store / f32 + residual store.
// ---------------------------------------------------------------------------
enum { EP_BF16 = 0, EP_SILUMUL = 1, EP_F32RES = 2 };

template <int EPI>
__global__ __launch_bounds__(256, 1) void gemm_bf16_kernel(
    const unsigned short* __restrict__ A,
    const unsigned short* __restrict__ B0,
    const unsigned short* __restrict__ B1,
    float* __restrict__ outF, unsigned short* __restrict__ outH,
    const float* __restrict__ resid,
    int M, int N, int K) {
    constexpr bool DUAL = (EPI == EP_SILUMUL);
    __shared__ unsigned short As[2][128 * RS];
    __shared__ unsigned short Bs[2][128 * RS];
    __shared__ unsigned short B1s[DUAL ? 2 : 1][DUAL ? 128 * RS : 1];

    int tid = threadIdx.x, lane = tid & 31, wv = tid >> 5;
    int l16 = lane & 15, lh = lane >> 4;
    int wm = wv >> 2, wn = wv & 3;
    int mBase = blockIdx.y * 128 + wm * 64;
    int nBase = blockIdx.x * 128 + wn * 32;
    (void)M;

    const unsigned short* Ag  = A  + (size_t)(blockIdx.y * 128) * K;
    const unsigned short* Bg  = B0 + (size_t)(blockIdx.x * 128) * K;
    const unsigned short* B1g = DUAL ? (B1 + (size_t)(blockIdx.x * 128) * K) : nullptr;

    v8f acc0[4][2];
    v8f acc1[4][2];
#pragma unroll
    for (int sm = 0; sm < 4; ++sm)
#pragma unroll
        for (int sn = 0; sn < 2; ++sn) {
            v8f z = {};
            acc0[sm][sn] = z;
            acc1[sm][sn] = z;
        }

    // prologue: stage k0 = 0 into buffer 0
    async_copy_tile(Ag, As[0], K, 0, tid);
    async_copy_tile(Bg, Bs[0], K, 0, tid);
    if constexpr (DUAL) async_copy_tile(B1g, B1s[0], K, 0, tid);

    for (int k0 = 0; k0 < K; k0 += 32) {
        int buf = (k0 >> 5) & 1;
        bool more = (k0 + 32 < K);                        // uniform
        if (more) {                                       // stage next tile
            async_copy_tile(Ag, As[buf ^ 1], K, k0 + 32, tid);
            async_copy_tile(Bg, Bs[buf ^ 1], K, k0 + 32, tid);
            if constexpr (DUAL)
                async_copy_tile(B1g, B1s[buf ^ 1], K, k0 + 32, tid);
            if constexpr (DUAL)
                asm volatile("s_wait_asynccnt 6" ::: "memory");
            else
                asm volatile("s_wait_asynccnt 4" ::: "memory");
        } else {
            asm volatile("s_wait_asynccnt 0" ::: "memory");
        }
        __syncthreads();                                  // tiles visible to all

        v16bf af[4], bf0[2], bf1[2];
#pragma unroll
        for (int sm = 0; sm < 4; ++sm)
            af[sm] = load_frag16(&As[buf][(wm * 64 + sm * 16 + l16) * RS + lh * 8]);
#pragma unroll
        for (int sn = 0; sn < 2; ++sn)
            bf0[sn] = load_frag16(&Bs[buf][(wn * 32 + sn * 16 + l16) * RS + lh * 8]);
        if constexpr (DUAL) {
#pragma unroll
            for (int sn = 0; sn < 2; ++sn)
                bf1[sn] = load_frag16(&B1s[buf][(wn * 32 + sn * 16 + l16) * RS + lh * 8]);
        }
#pragma unroll
        for (int sm = 0; sm < 4; ++sm)
#pragma unroll
            for (int sn = 0; sn < 2; ++sn)
                acc0[sm][sn] = wmma_bf16(af[sm], bf0[sn], acc0[sm][sn]);
        if constexpr (DUAL) {
#pragma unroll
            for (int sm = 0; sm < 4; ++sm)
#pragma unroll
                for (int sn = 0; sn < 2; ++sn)
                    acc1[sm][sn] = wmma_bf16(af[sm], bf1[sn], acc1[sm][sn]);
        }
        __syncthreads();                                  // done reading buf
    }

#pragma unroll
    for (int sm = 0; sm < 4; ++sm)
#pragma unroll
        for (int sn = 0; sn < 2; ++sn) {
            int gn = nBase + sn * 16 + l16;
#pragma unroll
            for (int j = 0; j < 8; ++j) {
                int gm = mBase + sm * 16 + lh * 8 + j;
                size_t idx = (size_t)gm * N + gn;
                float v = acc0[sm][sn][j];
                if constexpr (EPI == EP_BF16) {
                    outH[idx] = f2bf(v);
                } else if constexpr (EPI == EP_F32RES) {
                    outF[idx] = v + resid[idx];
                } else {
                    float sg = v / (1.f + __expf(-v));           // silu
                    outH[idx] = f2bf(sg * acc1[sm][sn][j]);
                }
            }
        }
}

// ---------------------------------------------------------------------------
// RoPE (positions 0..SEQ-1) applied in-place to bf16 q (32 heads) and k (8).
// blockIdx.y: 0..31 -> q head, 32..39 -> k head. 64 threads = 64 pairs.
// ---------------------------------------------------------------------------
__global__ void rope_kernel(unsigned short* __restrict__ qb,
                            unsigned short* __restrict__ kb,
                            const float* __restrict__ freqs) {
    int pos = blockIdx.x, h = blockIdx.y, p = threadIdx.x;
    float c = freqs[((size_t)pos * 64 + p) * 2 + 0];
    float s = freqs[((size_t)pos * 64 + p) * 2 + 1];
    unsigned short* ptr = (h < NH)
        ? (qb + (size_t)pos * DIM + h * HD + p * 2)
        : (kb + (size_t)pos * KVD + (h - NH) * HD + p * 2);
    float t0 = bf2f(ptr[0]), t1 = bf2f(ptr[1]);
    ptr[0] = f2bf(t0 * c - t1 * s);
    ptr[1] = f2bf(t0 * s + t1 * c);
}

// ---------------------------------------------------------------------------
// Shift cache by SEQ, append new (rope'd) k / v.  Emits fp32 cache outputs
// plus bf16 K-cache (pos,kvh,d) and bf16 transposed V (kvh,d,pos) for WMMA.
// ---------------------------------------------------------------------------
__global__ void cache_update_kernel(
    const float* __restrict__ ck_in, const float* __restrict__ cv_in,
    const unsigned short* __restrict__ kb, const unsigned short* __restrict__ vb,
    float* __restrict__ ck_out, float* __restrict__ cv_out,
    unsigned short* __restrict__ kc, unsigned short* __restrict__ vT) {
    int idx = blockIdx.x * 256 + threadIdx.x;   // CACHE*KVD elements
    int pos = idx >> 10, rem = idx & 1023;
    int kvh = rem >> 7, d = rem & 127;
    float kval, vval;
    if (pos < CACHE - SEQ) {
        kval = ck_in[(size_t)(pos + SEQ) * KVD + rem];
        vval = cv_in[(size_t)(pos + SEQ) * KVD + rem];
    } else {
        kval = bf2f(kb[(size_t)(pos - (CACHE - SEQ)) * KVD + rem]);
        vval = bf2f(vb[(size_t)(pos - (CACHE - SEQ)) * KVD + rem]);
    }
    ck_out[idx] = kval;
    cv_out[idx] = vval;
    kc[idx] = f2bf(kval);
    vT[((size_t)kvh * HD + d) * CACHE + pos] = f2bf(vval);
}

// ---------------------------------------------------------------------------
// Flash attention (GQA 4:1). Grid (SEQ/64, NH); 4 waves, one 16-row q tile
// each.  QK^T and PV on WMMA bf16; online softmax with 16-lane shfl_xor row
// reductions; P goes through LDS to re-form A fragments.
// ---------------------------------------------------------------------------
__global__ __launch_bounds__(128, 1) void attn_kernel(
    const unsigned short* __restrict__ q,
    const unsigned short* __restrict__ kc,
    const unsigned short* __restrict__ vT,
    unsigned short* __restrict__ attn_out) {
    int head = blockIdx.y, kvh = head >> 2;
    int wave = threadIdx.x >> 5, lane = threadIdx.x & 31;
    int l16 = lane & 15, lh = lane >> 4;
    int qg = blockIdx.x * 64;
    int q0 = qg + wave * 16;
    __shared__ unsigned short Pl[4][16][32];

    v16bf qf[4];
    const unsigned short* qrow = q + (size_t)(q0 + l16) * DIM + head * HD + lh * 8;
#pragma unroll
    for (int t = 0; t < 4; ++t) qf[t] = load_frag16(qrow + t * 32);

    v8f o[8];
    float mrow[8], lrow[8];
#pragma unroll
    for (int t = 0; t < 8; ++t) { v8f z = {}; o[t] = z; }
#pragma unroll
    for (int r = 0; r < 8; ++r) { mrow[r] = -1e30f; lrow[r] = 0.f; }

    // causal limit: j <= (CACHE-SEQ) + qi ; qi_max in group = qg+63
    int kvEnd = (CACHE - SEQ) + qg + 64;
    if (kvEnd > CACHE) kvEnd = CACHE;
    const float sc = 0.088388347648318447f;   // 1/sqrt(128)

    for (int kv0 = 0; kv0 < kvEnd; kv0 += 32) {
        v8f s[2];
        { v8f z = {}; s[0] = z; s[1] = z; }
#pragma unroll
        for (int sn = 0; sn < 2; ++sn) {
            const unsigned short* krow =
                kc + (size_t)(kv0 + sn * 16 + l16) * KVD + kvh * HD + lh * 8;
#pragma unroll
            for (int t = 0; t < 4; ++t)
                s[sn] = wmma_bf16(qf[t], load_frag16(krow + t * 32), s[sn]);
        }
#pragma unroll
        for (int r = 0; r < 8; ++r) {
            int qi = q0 + lh * 8 + r;
            float x0 = s[0][r] * sc, x1 = s[1][r] * sc;
            if (kv0 + l16 > (CACHE - SEQ) + qi) x0 = -1e9f;
            if (kv0 + 16 + l16 > (CACHE - SEQ) + qi) x1 = -1e9f;
            float loc = fmaxf(x0, x1);
#pragma unroll
            for (int off = 1; off < 16; off <<= 1)
                loc = fmaxf(loc, __shfl_xor(loc, off, 32));
            float mNew = fmaxf(mrow[r], loc);
            float corr = __expf(mrow[r] - mNew);
            float p0 = __expf(x0 - mNew), p1 = __expf(x1 - mNew);
            float rs = p0 + p1;
#pragma unroll
            for (int off = 1; off < 16; off <<= 1) rs += __shfl_xor(rs, off, 32);
            lrow[r] = lrow[r] * corr + rs;
            mrow[r] = mNew;
#pragma unroll
            for (int t = 0; t < 8; ++t) o[t][r] *= corr;
            Pl[wave][lh * 8 + r][l16] = f2bf(p0);
            Pl[wave][lh * 8 + r][16 + l16] = f2bf(p1);
        }
        __syncthreads();
        v8bf plo = *reinterpret_cast<const v8bf*>(&Pl[wave][l16][lh * 8]);
        v8bf phi = *reinterpret_cast<const v8bf*>(&Pl[wave][l16][lh * 8 + 16]);
        v16bf pf = __builtin_shufflevector(plo, phi, 0, 1, 2, 3, 4, 5, 6, 7,
                                           8, 9, 10, 11, 12, 13, 14, 15);
#pragma unroll
        for (int t = 0; t < 8; ++t) {
            const unsigned short* vrow =
                vT + ((size_t)kvh * HD + t * 16 + l16) * CACHE + kv0 + lh * 8;
            o[t] = wmma_bf16(pf, load_frag16(vrow), o[t]);
        }
        __syncthreads();
    }

#pragma unroll
    for (int t = 0; t < 8; ++t)
#pragma unroll
        for (int r = 0; r < 8; ++r) {
            int qi = q0 + lh * 8 + r;
            attn_out[(size_t)qi * DIM + head * HD + t * 16 + l16] =
                f2bf(o[t][r] / lrow[r]);
        }
}

// ---------------------------------------------------------------------------
// Host-side orchestration.
// ---------------------------------------------------------------------------
extern "C" void kernel_launch(void* const* d_in, const int* in_sizes, int n_in,
                              void* d_out, int out_size, void* d_ws, size_t ws_size,
                              hipStream_t stream) {
    (void)in_sizes; (void)n_in; (void)out_size; (void)ws_size;

    const float* x     = (const float*)d_in[0];
    /* d_in[1] = mask: applied analytically in attn_kernel */
    const float* freqs = (const float*)d_in[2];
    const float* ck_in = (const float*)d_in[3];
    const float* cv_in = (const float*)d_in[4];
    const float* wq    = (const float*)d_in[5];
    const float* wk    = (const float*)d_in[6];
    const float* wv    = (const float*)d_in[7];
    const float* wo    = (const float*)d_in[8];
    const float* w1    = (const float*)d_in[9];
    const float* w2    = (const float*)d_in[10];
    const float* w3    = (const float*)d_in[11];
    const float* anw   = (const float*)d_in[12];
    const float* fnw   = (const float*)d_in[13];

    float* out   = (float*)d_out;
    float* ckOut = out + (size_t)SEQ * DIM;
    float* cvOut = ckOut + (size_t)CACHE * KVD;

    char* ws = (char*)d_ws;
    size_t off = 0;
    auto alloc = [&](size_t bytes) -> void* {
        void* p = ws + off;
        off = (off + bytes + 255) & ~(size_t)255;
        return p;
    };
    typedef unsigned short u16;
    u16* wqT = (u16*)alloc(2ull * DIM * DIM);
    u16* wkT = (u16*)alloc(2ull * KVD * DIM);
    u16* wvT = (u16*)alloc(2ull * KVD * DIM);
    u16* woT = (u16*)alloc(2ull * DIM * DIM);
    u16* w1T = (u16*)alloc(2ull * HIDDEN * DIM);
    u16* w3T = (u16*)alloc(2ull * HIDDEN * DIM);
    u16* w2T = (u16*)alloc(2ull * DIM * HIDDEN);
    u16* xn  = (u16*)alloc(2ull * SEQ * DIM);
    u16* qb  = (u16*)alloc(2ull * SEQ * DIM);
    u16* kb  = (u16*)alloc(2ull * SEQ * KVD);
    u16* vb  = (u16*)alloc(2ull * SEQ * KVD);
    u16* kcb = (u16*)alloc(2ull * CACHE * KVD);
    u16* vTb = (u16*)alloc(2ull * CACHE * KVD);
    u16* ab  = (u16*)alloc(2ull * SEQ * DIM);     // attention output
    u16* hn  = (u16*)alloc(2ull * SEQ * DIM);
    u16* g   = (u16*)alloc(2ull * SEQ * HIDDEN);
    float* h = (float*)alloc(4ull * SEQ * DIM);

    dim3 tb(32, 8);
    // weight convert+transpose (fp32 KxN -> bf16 NxK); grid = (N/32, K/32)
    transpose_convert_kernel<<<dim3(DIM / 32, DIM / 32), tb, 0, stream>>>(wq, wqT, DIM, DIM);
    transpose_convert_kernel<<<dim3(KVD / 32, DIM / 32), tb, 0, stream>>>(wk, wkT, DIM, KVD);
    transpose_convert_kernel<<<dim3(KVD / 32, DIM / 32), tb, 0, stream>>>(wv, wvT, DIM, KVD);
    transpose_convert_kernel<<<dim3(DIM / 32, DIM / 32), tb, 0, stream>>>(wo, woT, DIM, DIM);
    transpose_convert_kernel<<<dim3(HIDDEN / 32, DIM / 32), tb, 0, stream>>>(w1, w1T, DIM, HIDDEN);
    transpose_convert_kernel<<<dim3(HIDDEN / 32, DIM / 32), tb, 0, stream>>>(w3, w3T, DIM, HIDDEN);
    transpose_convert_kernel<<<dim3(DIM / 32, HIDDEN / 32), tb, 0, stream>>>(w2, w2T, HIDDEN, DIM);

    // attn rmsnorm
    rmsnorm_bf16_kernel<<<SEQ, 256, 0, stream>>>(x, anw, xn);

    // QKV projections
    gemm_bf16_kernel<EP_BF16><<<dim3(DIM / 128, SEQ / 128), 256, 0, stream>>>(
        xn, wqT, nullptr, nullptr, qb, nullptr, SEQ, DIM, DIM);
    gemm_bf16_kernel<EP_BF16><<<dim3(KVD / 128, SEQ / 128), 256, 0, stream>>>(
        xn, wkT, nullptr, nullptr, kb, nullptr, SEQ, KVD, DIM);
    gemm_bf16_kernel<EP_BF16><<<dim3(KVD / 128, SEQ / 128), 256, 0, stream>>>(
        xn, wvT, nullptr, nullptr, vb, nullptr, SEQ, KVD, DIM);

    // RoPE + cache shift/append
    rope_kernel<<<dim3(SEQ, NH + NKV), 64, 0, stream>>>(qb, kb, freqs);
    cache_update_kernel<<<(CACHE * KVD) / 256, 256, 0, stream>>>(
        ck_in, cv_in, kb, vb, ckOut, cvOut, kcb, vTb);

    // flash attention
    attn_kernel<<<dim3(SEQ / 64, NH), 128, 0, stream>>>(qb, kcb, vTb, ab);

    // h = x + attn @ wo
    gemm_bf16_kernel<EP_F32RES><<<dim3(DIM / 128, SEQ / 128), 256, 0, stream>>>(
        ab, woT, nullptr, h, nullptr, x, SEQ, DIM, DIM);

    // ffn rmsnorm
    rmsnorm_bf16_kernel<<<SEQ, 256, 0, stream>>>(h, fnw, hn);

    // g = silu(hn@w1) * (hn@w3)   (fused dual-B GEMM)
    gemm_bf16_kernel<EP_SILUMUL><<<dim3(HIDDEN / 128, SEQ / 128), 256, 0, stream>>>(
        hn, w1T, w3T, nullptr, g, nullptr, SEQ, HIDDEN, DIM);

    // out = h + g @ w2
    gemm_bf16_kernel<EP_F32RES><<<dim3(DIM / 128, SEQ / 128), 256, 0, stream>>>(
        g, w2T, nullptr, out, nullptr, h, SEQ, DIM, HIDDEN);
}
